// NeRF_11845519802609
// MI455X (gfx1250) — matrix operand order repaired and checked by reading, not compiled, not run on hardware
//
#include <hip/hip_runtime.h>
#include <hip/hip_bf16.h>
#include <math.h>

typedef __attribute__((ext_vector_type(16))) _Float16 v16h;
typedef __attribute__((ext_vector_type(8)))  float    v8f;

#define TILE_M 64

static __device__ inline v8f wmma16(v16h a, v16h b, v8f c) {
  // D = A(16x32 f16) * B(32x16 f16) + C(16x16 f32)
  return __builtin_amdgcn_wmma_f32_16x16x32_f16(false, a, false, b, (short)0, c,
                                                false, false);
}

// ---- fragment loaders (wave32; A/B in contiguous k order, k = 16*(lane/16) + s) -----------------

// A fragment (16x32) from an f32 LDS tile (row-major, ld floats), optional relu
static __device__ inline v16h ldsA_f32(const float* xs, int ld, int r0, int k0,
                                       bool relu) {
  int lane = threadIdx.x & 31;
  int row  = r0 + (lane & 15);
  int kadd = (lane & 16) ? 16 : 0;
  v16h a;
#pragma unroll
  for (int h = 0; h < 16; ++h) {
    float f = xs[row * ld + k0 + h + kadd];
    if (relu) f = fmaxf(f, 0.f);
    a[h] = (_Float16)f;
  }
  return a;
}

// A fragment (16x32) from an f16 LDS tile
static __device__ inline v16h ldsA_f16(const _Float16* xs, int ld, int r0,
                                       int k0) {
  int lane = threadIdx.x & 31;
  int row  = r0 + (lane & 15);
  int kadd = (lane & 16) ? 16 : 0;
  v16h a;
#pragma unroll
  for (int h = 0; h < 16; ++h)
    a[h] = xs[row * ld + k0 + h + kadd];
  return a;
}

// B fragment (32x16) from global f16 weights stored TRANSPOSED [N][K] so that
// each lane reads 16 contiguous halves (32B, aligned) for its column.
static __device__ inline v16h gB(const _Float16* WT, int ldk, int n0, int k0) {
  int lane = threadIdx.x & 31;
  int col  = n0 + (lane & 15);
  int kb   = k0 + ((lane & 16) ? 16 : 0);
  return *(const v16h*)(WT + (size_t)col * ldk + kb);
}

// C (16x16 f32) store to f16 LDS tile, + bias, optional relu
static __device__ inline void stC_f16(_Float16* dst, int ld, int r0, int n0,
                                      v8f c, const float* bias, bool relu) {
  int lane = threadIdx.x & 31;
  int col  = n0 + (lane & 15);
  int radd = (lane & 16) ? 8 : 0;
  float bv = bias ? bias[col] : 0.f;
#pragma unroll
  for (int r = 0; r < 8; ++r) {
    float v = c[r] + bv;
    if (relu) v = fmaxf(v, 0.f);
    dst[(r0 + r + radd) * ld + col] = (_Float16)v;
  }
}

// C accumulate (+bias) into f32 LDS tile (residual add)
static __device__ inline void addC_f32(float* dst, int ld, int r0, int n0,
                                       v8f c, const float* bias) {
  int lane = threadIdx.x & 31;
  int col  = n0 + (lane & 15);
  int radd = (lane & 16) ? 8 : 0;
  float bv = bias[col];
#pragma unroll
  for (int r = 0; r < 8; ++r)
    dst[(r0 + r + radd) * ld + col] += c[r] + bv;
}

// ---- weight transpose + f32->f16 conversion ---------------------------------
// src: [K][Nc] f32 row-major; dst: [Nc][K] f16 row-major; blockIdx.y = batch
__global__ __launch_bounds__(256) void convT_kernel(const float* __restrict__ src,
                                                    _Float16* __restrict__ dst,
                                                    int K, int Nc) {
  int total   = K * Nc;
  size_t base = (size_t)blockIdx.y * total;
  int i = blockIdx.x * 256 + threadIdx.x;
  if (i >= total) return;
  int nr = i / K, k = i - nr * K;
  dst[base + i] = (_Float16)src[base + (size_t)k * Nc + nr];
}

// ---- point-independent small precompute -------------------------------------
__global__ __launch_bounds__(128) void prep_small_kernel(
    const float* emb, const float* pW, const float* pb,
    const float* Winp0, const float* binp0, const float* Wk0, const float* bk0,
    const float* Wv0, const float* bv0,
    const float* Winp1, const float* binp1, const float* Wk1, const float* bk1,
    const float* Wv1, const float* bv1,
    float* kmat, float* vmat) {
  __shared__ float pf[8 * 128];
  __shared__ float pp[8 * 128];
  const int t = threadIdx.x;
  for (int i = t; i < 1024; i += 128) {
    int r = i >> 7, c = i & 127;
    float a = pb[c];
    for (int k = 0; k < 128; ++k) a += emb[r * 128 + k] * pW[k * 128 + c];
    pf[i] = a;
  }
  __syncthreads();
  const float* Winp[2] = {Winp0, Winp1};
  const float* binp[2] = {binp0, binp1};
  const float* Wk[2]   = {Wk0, Wk1};
  const float* bk[2]   = {bk0, bk1};
  const float* Wv[2]   = {Wv0, Wv1};
  const float* bv[2]   = {bv0, bv1};
  for (int blk = 0; blk < 2; ++blk) {
    for (int i = t; i < 1024; i += 128) {
      int r = i >> 7, c = i & 127;
      float a = binp[blk][c];
      for (int k = 0; k < 128; ++k) a += pf[r * 128 + k] * Winp[blk][k * 128 + c];
      pp[i] = a;
    }
    __syncthreads();
    for (int i = t; i < 256; i += 128) {
      int r = i >> 5, c = i & 31;
      float a = bk[blk][c];
      for (int k = 0; k < 128; ++k) a += pp[r * 128 + k] * Wk[blk][k * 32 + c];
      kmat[blk * 256 + i] = a;
    }
    for (int i = t; i < 1024; i += 128) {
      int r = i >> 7, c = i & 127;
      float a = bv[blk][c];
      for (int k = 0; k < 128; ++k) a += pp[r * 128 + k] * Wv[blk][k * 128 + c];
      vmat[blk * 1024 + i] = a;
    }
    __syncthreads();
  }
}

// ---- view-embedding feature build: [k0(12), dirs(3), sin(12), cos(12)] ------
__global__ __launch_bounds__(256) void build_feats_kernel(
    const float* __restrict__ k0, const float* __restrict__ dirs,
    float* __restrict__ feats, int n) {
  int i = blockIdx.x * 256 + threadIdx.x;
  if (i >= n) return;
  float* f = feats + (size_t)i * 40;
  const float* kk = k0 + (size_t)i * 12;
#pragma unroll
  for (int j = 0; j < 12; ++j) f[j] = kk[j];
  float d0 = dirs[i * 3 + 0], d1 = dirs[i * 3 + 1], d2 = dirs[i * 3 + 2];
  f[12] = d0; f[13] = d1; f[14] = d2;
  float dd[3] = {d0, d1, d2};
#pragma unroll
  for (int c = 0; c < 3; ++c)
#pragma unroll
    for (int pe = 0; pe < 4; ++pe) {
      float v = dd[c] * (float)(1 << pe);
      f[15 + c * 4 + pe] = __sinf(v);
      f[27 + c * 4 + pe] = __cosf(v);
    }
  f[39] = 0.f;
}

// ---- fused cross-attention MLP block ----------------------------------------
struct MlpP {
  const float* xin; int din; int xin_ld;
  const float* Win; const float* bin;
  const _Float16* WqT; const float* bq;
  const float* kmat; const float* vmat;
  const float* g1; const float* be1;
  const _Float16* Wf1T; const float* bf1;
  const _Float16* Wf2T; const float* bf2;
  const float* g2; const float* be2;
  const _Float16* WoutT; const float* bout; int dout;
  float* out; int out_ld;
};

__global__ __launch_bounds__(128) void mlp_block_kernel(MlpP P) {
  __shared__ float    xs[TILE_M * 128];   // f32 activations (residual/LN)
  __shared__ _Float16 hb[TILE_M * 256];   // f16 hidden tile (also q staging)
  const int t  = threadIdx.x;
  const int m0 = blockIdx.x * TILE_M;
  const int r0 = (t >> 5) * 16;           // wave's row stripe

  // (a) x = xin @ Win + bin  (K = din <= 39, scalar)
  {
    int row = t >> 1, cb = (t & 1) * 64;
    const float* xr = P.xin + (size_t)(m0 + row) * P.xin_ld;
    for (int c = 0; c < 64; ++c) {
      int col = cb + c;
      float acc = P.bin[col];
      for (int k = 0; k < P.din; ++k) acc += xr[k] * P.Win[k * 128 + col];
      xs[row * 128 + col] = acc;
    }
  }
  __syncthreads();

  // (b) q = x @ Wq + bq  -> hb[:,0:32] as f16 (WMMA); A frags hoisted
  {
    v16h a[4];
#pragma unroll
    for (int kt = 0; kt < 4; ++kt) a[kt] = ldsA_f32(xs, 128, r0, kt * 32, false);
    for (int nt = 0; nt < 2; ++nt) {
      v8f c = {};
#pragma unroll
      for (int kt = 0; kt < 4; ++kt)
        c = wmma16(a[kt], gB(P.WqT, 128, nt * 16, kt * 32), c);
      stC_f16(hb, 256, r0, nt * 16, c, P.bq, false);
    }
  }
  __syncthreads();

  // (c) attention over P=8 parts + residual + LN1 (one thread per row)
  if (t < TILE_M) {
    const float scale = 0.1767766952966369f;  // 32^-0.5
    float sum = 0.f, ssq = 0.f;
    for (int h2 = 0; h2 < 4; ++h2) {
      float att[8]; float mx = -3.0e38f;
      for (int p = 0; p < 8; ++p) {
        float l = 0.f;
        for (int d = 0; d < 8; ++d)
          l += (float)hb[t * 256 + h2 * 8 + d] * P.kmat[p * 32 + h2 * 8 + d];
        l *= scale; att[p] = l; mx = fmaxf(mx, l);
      }
      float s = 0.f;
      for (int p = 0; p < 8; ++p) { att[p] = __expf(att[p] - mx); s += att[p]; }
      float inv = 1.f / s;
      for (int e = 0; e < 32; ++e) {
        float o = 0.f;
        for (int p = 0; p < 8; ++p) o += att[p] * P.vmat[p * 128 + h2 * 32 + e];
        int c = h2 * 32 + e;
        float v = xs[t * 128 + c] + o * inv;
        xs[t * 128 + c] = v; sum += v; ssq += v * v;
      }
    }
    float mean = sum * (1.f / 128.f);
    float inv  = rsqrtf(ssq * (1.f / 128.f) - mean * mean + 1e-5f);
    for (int c = 0; c < 128; ++c)
      xs[t * 128 + c] = (xs[t * 128 + c] - mean) * inv * P.g1[c] + P.be1[c];
  }
  __syncthreads();

  // (d) hidden = relu(x @ Wf1 + bf1) -> hb [64][256]; A frags hoisted
  {
    v16h a[4];
#pragma unroll
    for (int kt = 0; kt < 4; ++kt) a[kt] = ldsA_f32(xs, 128, r0, kt * 32, false);
    for (int nt = 0; nt < 16; ++nt) {
      v8f c = {};
#pragma unroll
      for (int kt = 0; kt < 4; ++kt)
        c = wmma16(a[kt], gB(P.Wf1T, 128, nt * 16, kt * 32), c);
      stC_f16(hb, 256, r0, nt * 16, c, P.bf1, true);
    }
  }
  __syncthreads();

  // (e) x += hidden @ Wf2 + bf2 (K=256); A frags hoisted (8 frags)
  {
    v16h a[8];
#pragma unroll
    for (int kt = 0; kt < 8; ++kt) a[kt] = ldsA_f16(hb, 256, r0, kt * 32);
    for (int nt = 0; nt < 8; ++nt) {
      v8f c = {};
#pragma unroll
      for (int kt = 0; kt < 8; ++kt)
        c = wmma16(a[kt], gB(P.Wf2T, 256, nt * 16, kt * 32), c);
      addC_f32(xs, 128, r0, nt * 16, c, P.bf2);
    }
  }
  __syncthreads();

  // LN2
  if (t < TILE_M) {
    float sum = 0.f, ssq = 0.f;
    for (int c = 0; c < 128; ++c) { float v = xs[t * 128 + c]; sum += v; ssq += v * v; }
    float mean = sum * (1.f / 128.f);
    float inv  = rsqrtf(ssq * (1.f / 128.f) - mean * mean + 1e-5f);
    for (int c = 0; c < 128; ++c)
      xs[t * 128 + c] = (xs[t * 128 + c] - mean) * inv * P.g2[c] + P.be2[c];
  }
  __syncthreads();

  // (f) out = relu(x) @ Wout + bout -> global; A frags hoisted
  {
    v16h a[4];
#pragma unroll
    for (int kt = 0; kt < 4; ++kt) a[kt] = ldsA_f32(xs, 128, r0, kt * 32, true);
    int ntiles = (P.dout + 15) >> 4;
    for (int nt = 0; nt < ntiles; ++nt) {
      v8f c = {};
#pragma unroll
      for (int kt = 0; kt < 4; ++kt)
        c = wmma16(a[kt], gB(P.WoutT, 128, nt * 16, kt * 32), c);
      int lane = t & 31;
      int col  = nt * 16 + (lane & 15);
      int radd = (lane & 16) ? 8 : 0;
      if (col < P.dout) {
        float bv = P.bout[col];
#pragma unroll
        for (int r = 0; r < 8; ++r)
          P.out[(size_t)(m0 + r0 + r + radd) * P.out_ld + col] = c[r] + bv;
      }
    }
  }
}

// ---- 8-expert rgbnets, gate-mixed, sigmoid ----------------------------------
__global__ __launch_bounds__(128) void experts_kernel(
    const float* __restrict__ hsrc, const float* __restrict__ logits,
    const _Float16* __restrict__ W1T, const float* __restrict__ b1,
    const _Float16* __restrict__ W2T, const float* __restrict__ b2,
    const float* __restrict__ W3, const float* __restrict__ b3,
    float* __restrict__ out) {
  __shared__ _Float16 hs[TILE_M * 128];
  __shared__ _Float16 e1[TILE_M * 128];
  __shared__ _Float16 e2[TILE_M * 128];
  __shared__ float gate[TILE_M * 8];
  __shared__ float acc[TILE_M * 3];
  const int t  = threadIdx.x;
  const int m0 = blockIdx.x * TILE_M;
  const int r0 = (t >> 5) * 16;

  for (int i = t; i < TILE_M * 128; i += 128)
    hs[i] = (_Float16)hsrc[(size_t)m0 * 128 + i];
  for (int i = t; i < TILE_M * 3; i += 128) acc[i] = 0.f;
  if (t < TILE_M) {  // gate = softmax(logits)
    float l[8], mx = -3.0e38f;
    for (int p = 0; p < 8; ++p) {
      l[p] = logits[(size_t)(m0 + t) * 8 + p];
      mx = fmaxf(mx, l[p]);
    }
    float s = 0.f;
    for (int p = 0; p < 8; ++p) { l[p] = __expf(l[p] - mx); s += l[p]; }
    float inv = 1.f / s;
    for (int p = 0; p < 8; ++p) gate[t * 8 + p] = l[p] * inv;
  }
  __syncthreads();

  // h A-fragments are invariant across all experts: load once.
  v16h ah[4];
#pragma unroll
  for (int kt = 0; kt < 4; ++kt) ah[kt] = ldsA_f16(hs, 128, r0, kt * 32);

  for (int p = 0; p < 8; ++p) {
    const _Float16* w1 = W1T + (size_t)p * 128 * 128;
    const _Float16* w2 = W2T + (size_t)p * 128 * 128;
    for (int nt = 0; nt < 8; ++nt) {
      v8f c = {};
#pragma unroll
      for (int kt = 0; kt < 4; ++kt)
        c = wmma16(ah[kt], gB(w1, 128, nt * 16, kt * 32), c);
      stC_f16(e1, 128, r0, nt * 16, c, b1 + p * 128, true);
    }
    __syncthreads();
    {
      v16h ae[4];
#pragma unroll
      for (int kt = 0; kt < 4; ++kt) ae[kt] = ldsA_f16(e1, 128, r0, kt * 32);
      for (int nt = 0; nt < 8; ++nt) {
        v8f c = {};
#pragma unroll
        for (int kt = 0; kt < 4; ++kt)
          c = wmma16(ae[kt], gB(w2, 128, nt * 16, kt * 32), c);
        stC_f16(e2, 128, r0, nt * 16, c, b2 + p * 128, true);
      }
    }
    __syncthreads();
    if (t < TILE_M) {  // head 128->3, gate-mix
      float g = gate[t * 8 + p];
      for (int c = 0; c < 3; ++c) {
        float d = b3[p * 3 + c];
        for (int k = 0; k < 128; ++k)
          d += (float)e2[t * 128 + k] * W3[p * 384 + k * 3 + c];
        acc[t * 3 + c] += g * d;
      }
    }
    __syncthreads();
  }
  if (t < TILE_M)
    for (int c = 0; c < 3; ++c) {
      float v = acc[t * 3 + c];
      out[(size_t)(m0 + t) * 3 + c] = 1.f / (1.f + __expf(-v));
    }
}

// ---- launcher ---------------------------------------------------------------
extern "C" void kernel_launch(void* const* d_in, const int* in_sizes, int n_in,
                              void* d_out, int out_size, void* d_ws,
                              size_t ws_size, hipStream_t stream) {
  const float* k0v  = (const float*)d_in[0];
  const float* dirs = (const float*)d_in[1];
  const float* pemb = (const float*)d_in[2];
  const float* pW   = (const float*)d_in[3];
  const float* pb   = (const float*)d_in[4];
  // index_mlp
  const float* iWin  = (const float*)d_in[5];
  const float* iBin  = (const float*)d_in[6];
  const float* iWinp = (const float*)d_in[7];
  const float* iBinp = (const float*)d_in[8];
  const float* iWq   = (const float*)d_in[9];
  const float* iBq   = (const float*)d_in[10];
  const float* iWk   = (const float*)d_in[11];
  const float* iBk   = (const float*)d_in[12];
  const float* iWv   = (const float*)d_in[13];
  const float* iBv   = (const float*)d_in[14];
  const float* iWf1  = (const float*)d_in[15];
  const float* iBf1  = (const float*)d_in[16];
  const float* iWf2  = (const float*)d_in[17];
  const float* iBf2  = (const float*)d_in[18];
  const float* iG1   = (const float*)d_in[19];
  const float* iBe1  = (const float*)d_in[20];
  const float* iG2   = (const float*)d_in[21];
  const float* iBe2  = (const float*)d_in[22];
  const float* iWout = (const float*)d_in[23];
  const float* iBout = (const float*)d_in[24];
  // feat_mlp
  const float* fWin  = (const float*)d_in[25];
  const float* fBin  = (const float*)d_in[26];
  const float* fWinp = (const float*)d_in[27];
  const float* fBinp = (const float*)d_in[28];
  const float* fWq   = (const float*)d_in[29];
  const float* fBq   = (const float*)d_in[30];
  const float* fWk   = (const float*)d_in[31];
  const float* fBk   = (const float*)d_in[32];
  const float* fWv   = (const float*)d_in[33];
  const float* fBv   = (const float*)d_in[34];
  const float* fWf1  = (const float*)d_in[35];
  const float* fBf1  = (const float*)d_in[36];
  const float* fWf2  = (const float*)d_in[37];
  const float* fBf2  = (const float*)d_in[38];
  const float* fG1   = (const float*)d_in[39];
  const float* fBe1  = (const float*)d_in[40];
  const float* fG2   = (const float*)d_in[41];
  const float* fBe2  = (const float*)d_in[42];
  const float* fWout = (const float*)d_in[43];
  const float* fBout = (const float*)d_in[44];
  // rgbnets
  const float* rW1 = (const float*)d_in[45];
  const float* rB1 = (const float*)d_in[46];
  const float* rW2 = (const float*)d_in[47];
  const float* rB2 = (const float*)d_in[48];
  const float* rW3 = (const float*)d_in[49];
  const float* rB3 = (const float*)d_in[50];

  const int N = in_sizes[0] / 12;

  char* w = (char*)d_ws;
  size_t off = 0;
  auto alloc = [&](size_t bytes) -> char* {
    char* p = w + off;
    off += (bytes + 255) & ~(size_t)255;
    return p;
  };
  float* feats  = (float*)alloc((size_t)N * 40 * 4);
  float* logits = (float*)alloc((size_t)N * 8 * 4);
  float* hbuf   = (float*)alloc((size_t)N * 128 * 4);
  float* kmat   = (float*)alloc(2 * 8 * 32 * 4);
  float* vmat   = (float*)alloc(2 * 8 * 128 * 4);
  _Float16* WqT_i   = (_Float16*)alloc(32 * 128 * 2);
  _Float16* WqT_f   = (_Float16*)alloc(32 * 128 * 2);
  _Float16* Wf1T_i  = (_Float16*)alloc(256 * 128 * 2);
  _Float16* Wf1T_f  = (_Float16*)alloc(256 * 128 * 2);
  _Float16* Wf2T_i  = (_Float16*)alloc(128 * 256 * 2);
  _Float16* Wf2T_f  = (_Float16*)alloc(128 * 256 * 2);
  _Float16* WoutT_i = (_Float16*)alloc(8 * 128 * 2);
  _Float16* WoutT_f = (_Float16*)alloc(128 * 128 * 2);
  _Float16* W1T     = (_Float16*)alloc((size_t)8 * 128 * 128 * 2);
  _Float16* W2T     = (_Float16*)alloc((size_t)8 * 128 * 128 * 2);

  auto launchT = [&](const float* src, _Float16* dst, int K, int Nc, int batch) {
    dim3 g((unsigned)((K * Nc + 255) / 256), (unsigned)batch);
    convT_kernel<<<g, dim3(256), 0, stream>>>(src, dst, K, Nc);
  };
  launchT(iWq, WqT_i, 128, 32, 1);
  launchT(fWq, WqT_f, 128, 32, 1);
  launchT(iWf1, Wf1T_i, 128, 256, 1);
  launchT(fWf1, Wf1T_f, 128, 256, 1);
  launchT(iWf2, Wf2T_i, 256, 128, 1);
  launchT(fWf2, Wf2T_f, 256, 128, 1);
  launchT(iWout, WoutT_i, 128, 8, 1);
  launchT(fWout, WoutT_f, 128, 128, 1);
  launchT(rW1, W1T, 128, 128, 8);
  launchT(rW2, W2T, 128, 128, 8);

  prep_small_kernel<<<dim3(1), dim3(128), 0, stream>>>(
      pemb, pW, pb, iWinp, iBinp, iWk, iBk, iWv, iBv, fWinp, fBinp, fWk, fBk,
      fWv, fBv, kmat, vmat);

  build_feats_kernel<<<dim3((N + 255) / 256), dim3(256), 0, stream>>>(
      k0v, dirs, feats, N);

  MlpP Pi;
  Pi.xin = k0v;  Pi.din = 12; Pi.xin_ld = 12;
  Pi.Win = iWin; Pi.bin = iBin;
  Pi.WqT = WqT_i; Pi.bq = iBq;
  Pi.kmat = kmat; Pi.vmat = vmat;
  Pi.g1 = iG1; Pi.be1 = iBe1;
  Pi.Wf1T = Wf1T_i; Pi.bf1 = iBf1;
  Pi.Wf2T = Wf2T_i; Pi.bf2 = iBf2;
  Pi.g2 = iG2; Pi.be2 = iBe2;
  Pi.WoutT = WoutT_i; Pi.bout = iBout; Pi.dout = 8;
  Pi.out = logits; Pi.out_ld = 8;
  mlp_block_kernel<<<dim3(N / TILE_M), dim3(128), 0, stream>>>(Pi);

  MlpP Pf;
  Pf.xin = feats; Pf.din = 39; Pf.xin_ld = 40;
  Pf.Win = fWin; Pf.bin = fBin;
  Pf.WqT = WqT_f; Pf.bq = fBq;
  Pf.kmat = kmat + 256; Pf.vmat = vmat + 1024;
  Pf.g1 = fG1; Pf.be1 = fBe1;
  Pf.Wf1T = Wf1T_f; Pf.bf1 = fBf1;
  Pf.Wf2T = Wf2T_f; Pf.bf2 = fBf2;
  Pf.g2 = fG2; Pf.be2 = fBe2;
  Pf.WoutT = WoutT_f; Pf.bout = fBout; Pf.dout = 128;
  Pf.out = hbuf; Pf.out_ld = 128;
  mlp_block_kernel<<<dim3(N / TILE_M), dim3(128), 0, stream>>>(Pf);

  experts_kernel<<<dim3(N / TILE_M), dim3(128), 0, stream>>>(
      hbuf, logits, W1T, rB1, W2T, rB2, rW3, rB3, (float*)d_out);
}
